// TransformerDecoder_1554778161683
// MI455X (gfx1250) — matrix operand order repaired and checked by reading, not compiled, not run on hardware
//
#include <hip/hip_runtime.h>
#include <hip/hip_bf16.h>
#include <math.h>

// ---------------------------------------------------------------------------
// Transformer decoder forward (8 layers) for MI455X / gfx1250.
// GEMMs + attention via v_wmma_f32_16x16x32_bf16 (wave32 WMMA).
// A-tile staging via TDM tensor_load_to_lds; Q/K staging via
// global_load_async_to_lds_b128 (ASYNCcnt).
// ---------------------------------------------------------------------------

typedef __attribute__((ext_vector_type(8)))  float    v8f;
typedef __attribute__((ext_vector_type(16))) __bf16   v16bf;
typedef __attribute__((ext_vector_type(8)))  __bf16   v8bf;
typedef __attribute__((ext_vector_type(4)))  unsigned u32x4;
typedef __attribute__((ext_vector_type(8)))  unsigned u32x8;

#define LAYERS 8
#define BSZ    4
#define SEQ    1024
#define DIM    1024
#define HEADS  16
#define DHEAD  64
#define FCSZ   4096
#define MROWS  (BSZ*SEQ)

// ---------------------------------------------------------------------------
// WMMA fragment helpers (layouts per CDNA5 ISA 7.12.2, wave32)
// ---------------------------------------------------------------------------
__device__ __forceinline__ v16bf frag_combine(v8bf lo, v8bf hi) {
  v16bf r;
#pragma unroll
  for (int i = 0; i < 8; ++i) { r[i] = lo[i]; r[i + 8] = hi[i]; }
  return r;
}

// A fragment 16x32 bf16: lanes 0-15 row=lane, k-chunks {0..7, 16..23};
// lanes 16-31 row=lane-16, k-chunks {8..15, 24..31}. Two ds_load_b128 per lane.
__device__ __forceinline__ v16bf load_frag_a(const __bf16* base, int stride) {
  int lane = threadIdx.x & 31;
  int row  = lane & 15;
  int lo   = (lane >> 4) << 3;            // 0 or 8
  const __bf16* p = base + row * stride + lo;
  v8bf c0 = *(const v8bf*)(p);
  v8bf c1 = *(const v8bf*)(p + 16);
  return frag_combine(c0, c1);
}

// B fragment 32x16 bf16 loaded from a tile stored TRANSPOSED in LDS as [n][k]:
// lanes 0-15 col n=lane, k-chunks {0..7, 16..23}; lanes 16-31 col n=lane-16, k-chunks {8..15, 24..31}.
__device__ __forceinline__ v16bf load_frag_b(const __bf16* baseT, int stride) {
  int lane = threadIdx.x & 31;
  int n    = lane & 15;
  int ko   = (lane >> 4) << 3;            // 0 or 8
  const __bf16* p = baseT + n * stride + ko;
  v8bf c0 = *(const v8bf*)(p);
  v8bf c1 = *(const v8bf*)(p + 16);
  return frag_combine(c0, c1);
}

// LDS byte offset of a generic pointer into __shared__ (aperture truncation).
__device__ __forceinline__ unsigned lds_off(const void* p) {
  return (unsigned)(unsigned long long)(uintptr_t)p;
}

// Per-lane async global->LDS copy of 16 bytes (tracked by ASYNCcnt).
__device__ __forceinline__ void async_copy_b128(const void* gptr, const void* lptr) {
  asm volatile("global_load_async_to_lds_b128 %0, %1, off"
               :: "v"(lds_off(lptr)), "v"((unsigned long long)(uintptr_t)gptr)
               : "memory");
}

// ---------------------------------------------------------------------------
// f32 -> bf16 conversion
// ---------------------------------------------------------------------------
__global__ void f32_to_bf16_k(const float* __restrict__ in,
                              __bf16* __restrict__ out, int n) {
  int i = blockIdx.x * blockDim.x + threadIdx.x;
  if (i < n) out[i] = (__bf16)in[i];
}

// ---------------------------------------------------------------------------
// LayerNorm (shared gamma/beta, eps=1e-5). One block per row of D=1024.
// ---------------------------------------------------------------------------
template <typename OT>
__global__ __launch_bounds__(256)
void layernorm_k(const float* __restrict__ x, const float* __restrict__ g,
                 const float* __restrict__ b, OT* __restrict__ y, int D) {
  __shared__ float s1[256], s2[256];
  int row = blockIdx.x, t = threadIdx.x;
  const float* xr = x + (size_t)row * D;
  float a = 0.f, q = 0.f;
  for (int j = t; j < D; j += 256) { float v = xr[j]; a += v; q += v * v; }
  s1[t] = a; s2[t] = q;
  __syncthreads();
  for (int off = 128; off > 0; off >>= 1) {
    if (t < off) { s1[t] += s1[t + off]; s2[t] += s2[t + off]; }
    __syncthreads();
  }
  float mu   = s1[0] / (float)D;
  float var  = s2[0] / (float)D - mu * mu;
  float rstd = rsqrtf(var + 1e-5f);
  OT* yr = y + (size_t)row * D;
  for (int j = t; j < D; j += 256) {
    float v = (xr[j] - mu) * rstd * g[j] + b[j];
    yr[j] = (OT)v;
  }
}

// ---------------------------------------------------------------------------
// WMMA GEMM: C[M,N] = A[M,K] (bf16) @ W[K,N] (bf16) + bias, fused epilogue.
// MODE 0: -> bf16 out.  MODE 1: GELU(exact) -> bf16 out.
// MODE 2: xout = xin + scale * (acc + bias) -> f32 (re-zero residual).
// Block tile 128x128x32, 256 threads = 8 waves in a 4x2 wave grid.
// A tile staged by the Tensor Data Mover (one tensor_load_to_lds per K step,
// issued by wave 0, padded 64B data + 16B pad per row -> LSTR=40 elems).
// ---------------------------------------------------------------------------
#define BM 128
#define BN 128
#define BK 32
#define LSTR 40   // padded LDS row stride (elems); 80B rows -> conflict-free b128

template <int MODE>
__global__ __launch_bounds__(256)
void gemm_wmma(const __bf16* __restrict__ A, const __bf16* __restrict__ W,
               const float* __restrict__ bias, int M, int N, int K,
               __bf16* __restrict__ outB,
               const float* __restrict__ xin, float* __restrict__ xout,
               const float* __restrict__ scale_p) {
  __shared__ __bf16 As[BM * LSTR];
  __shared__ __bf16 Bs[BN * LSTR];

  const int t    = threadIdx.x;
  const int lane = t & 31;
  const int w    = t >> 5;      // 0..7
  const int wm   = w >> 1;      // 0..3 -> 32-row strip
  const int wn   = w & 1;       // 0..1 -> 64-col strip
  const int m0   = blockIdx.y * BM;
  const int n0   = blockIdx.x * BN;

  v8f acc[2][4];
#pragma unroll
  for (int mi = 0; mi < 2; ++mi)
#pragma unroll
    for (int ni = 0; ni < 4; ++ni)
#pragma unroll
      for (int r = 0; r < 8; ++r) acc[mi][ni][r] = 0.0f;

  // TDM descriptor pieces that do not change across the K loop.
  // D# group1: data_size=2B(code1), pad_enable, pad_interval=16DW(code3),
  //            pad_amount=4DW(code3); tensor dims [K, M]; tile [32, 128].
  const unsigned g1w0 = (1u << 16) | (1u << 20) | (3u << 22) | (3u << 25);
  const unsigned asOff = lds_off(&As[0]);

  for (int k0 = 0; k0 < K; k0 += BK) {
    // ---- A tile 128x32 via Tensor Data Mover (wave 0 issues) ----
    if (w == 0) {
      unsigned long long ga = (unsigned long long)(uintptr_t)A +
                              ((unsigned long long)(unsigned)m0 * (unsigned)K +
                               (unsigned)k0) * 2ull;
      u32x4 g0;
      g0[0] = 1u;                                   // count=1, user mode
      g0[1] = asOff;                                // lds_addr
      g0[2] = (unsigned)ga;                         // global_addr[31:0]
      g0[3] = ((unsigned)(ga >> 32) & 0x1FFFFFFu) | 0x80000000u; // addr hi | type=2
      u32x8 g1;
      g1[0] = g1w0;
      g1[1] = ((unsigned)K & 0xFFFFu) << 16;        // tensor_dim0 lo16
      g1[2] = (((unsigned)K >> 16) & 0xFFFFu) | (((unsigned)M & 0xFFFFu) << 16);
      g1[3] = (((unsigned)M >> 16) & 0xFFFFu) | (32u << 16);  // tile_dim0=32
      g1[4] = 128u;                                 // tile_dim1=128, tile_dim2=0
      g1[5] = (unsigned)K;                          // dim0_stride lo32
      g1[6] = 0u;
      g1[7] = 0u;
      asm volatile("tensor_load_to_lds %0, %1" :: "s"(g0), "s"(g1) : "memory");
      __builtin_amdgcn_s_wait_tensorcnt(0);
    }
    // ---- B tile 32x128 read coalesced, stored transposed [n][k] ----
#pragma unroll
    for (int i = 0; i < 2; ++i) {
      int c = t + i * 256;                      // 512 chunks of 8 elems
      int kk = c >> 4, ch = c & 15;
      const __bf16* gp = W + (size_t)(k0 + kk) * N + n0 + ch * 8;
      v8bf v = *(const v8bf*)gp;
#pragma unroll
      for (int e = 0; e < 8; ++e) Bs[(ch * 8 + e) * LSTR + kk] = v[e];
    }
    if (k0 + BK < K) {                          // global_prefetch_b8 next W tile
      __builtin_prefetch((const void*)(W + (size_t)(k0 + BK + (t >> 3)) * N + n0), 0, 1);
    }
    __syncthreads();

    v16bf af[2], bf[4];
#pragma unroll
    for (int mi = 0; mi < 2; ++mi)
      af[mi] = load_frag_a(&As[(wm * 32 + mi * 16) * LSTR], LSTR);
#pragma unroll
    for (int ni = 0; ni < 4; ++ni)
      bf[ni] = load_frag_b(&Bs[(wn * 64 + ni * 16) * LSTR], LSTR);
#pragma unroll
    for (int mi = 0; mi < 2; ++mi)
#pragma unroll
      for (int ni = 0; ni < 4; ++ni)
        acc[mi][ni] = __builtin_amdgcn_wmma_f32_16x16x32_bf16(
            false, af[mi], false, bf[ni], (short)0, acc[mi][ni], false, false);
    __syncthreads();
  }

  // ---- epilogue (C layout: N = lane&15, M = base + r + 8*(lane>=16)) ----
  float sc = (MODE == 2) ? scale_p[0] : 0.0f;
#pragma unroll
  for (int mi = 0; mi < 2; ++mi) {
    int mbase = m0 + wm * 32 + mi * 16 + ((lane >> 4) << 3);
#pragma unroll
    for (int ni = 0; ni < 4; ++ni) {
      int n = n0 + wn * 64 + ni * 16 + (lane & 15);
      float bn = bias[n];
#pragma unroll
      for (int r = 0; r < 8; ++r) {
        float v = acc[mi][ni][r] + bn;
        size_t idx = (size_t)(mbase + r) * N + n;
        if (MODE == 0) {
          outB[idx] = (__bf16)v;
        } else if (MODE == 1) {
          float gelu = 0.5f * v * (1.0f + erff(v * 0.70710678118654752f));
          outB[idx] = (__bf16)gelu;
        } else {
          xout[idx] = xin[idx] + sc * v;
        }
      }
    }
  }
}

// ---------------------------------------------------------------------------
// Flash attention (causal). Grid: (SEQ/64, BSZ*HEADS). 128 threads = 4 waves,
// each wave owns 16 q rows. Q/K tiles staged via async global->LDS copies.
// ---------------------------------------------------------------------------
#define ASTR 72

__global__ __launch_bounds__(128)
void attn_flash(const __bf16* __restrict__ Q, const __bf16* __restrict__ Km,
                const __bf16* __restrict__ Vm, __bf16* __restrict__ O) {
  __shared__ __bf16 Qs[64 * ASTR];
  __shared__ __bf16 Ks[64 * ASTR];
  __shared__ __bf16 Vt[64 * ASTR];       // V transposed: [dh][key]
  __shared__ __bf16 Ps[4 * 16 * ASTR];   // per-wave P staging

  const int t    = threadIdx.x;
  const int lane = t & 31;
  const int w    = t >> 5;               // 0..3
  const int qb   = blockIdx.x;
  const int bh   = blockIdx.y;
  const int b    = bh / HEADS, h = bh % HEADS;
  const size_t base = (size_t)b * SEQ * DIM + (size_t)h * DHEAD;

  // cooperative Q block load: 64 rows x 64 dh, async to LDS
#pragma unroll
  for (int i = 0; i < 4; ++i) {
    int c = t + i * 128;                 // 512 chunks of 8 elems
    int row = c >> 3, ch = c & 7;
    async_copy_b128(Q + base + (size_t)(qb * 64 + row) * DIM + ch * 8,
                    &Qs[row * ASTR + ch * 8]);
  }
  asm volatile("s_wait_asynccnt 0" ::: "memory");
  __syncthreads();

  v16bf aq0 = load_frag_a(&Qs[(w * 16) * ASTR], ASTR);        // dh 0..31
  v16bf aq1 = load_frag_a(&Qs[(w * 16) * ASTR + 32], ASTR);   // dh 32..63

  float mrow[8], lrow[8];
#pragma unroll
  for (int r = 0; r < 8; ++r) { mrow[r] = -3.0e38f; lrow[r] = 0.f; }
  v8f o[4];
#pragma unroll
  for (int ni = 0; ni < 4; ++ni)
#pragma unroll
    for (int r = 0; r < 8; ++r) o[ni][r] = 0.f;

  const int rhalf  = ((lane >> 4) << 3);
  const int qrow0  = qb * 64 + w * 16 + rhalf;
  __bf16* Pw = &Ps[(w * 16) * ASTR];

  for (int kb = 0; kb <= qb; ++kb) {
    __syncthreads();   // protect Ks/Vt against previous iteration's readers
    // K block -> Ks[key][dh], async to LDS
#pragma unroll
    for (int i = 0; i < 4; ++i) {
      int c = t + i * 128;
      int row = c >> 3, ch = c & 7;
      async_copy_b128(Km + base + (size_t)(kb * 64 + row) * DIM + ch * 8,
                      &Ks[row * ASTR + ch * 8]);
    }
    // V block -> Vt[dh][key] (transposed scatter through VGPRs)
#pragma unroll
    for (int i = 0; i < 4; ++i) {
      int c = t + i * 128;
      int row = c >> 3, ch = c & 7;
      v8bf v = *(const v8bf*)(Vm + base + (size_t)(kb * 64 + row) * DIM + ch * 8);
#pragma unroll
      for (int e = 0; e < 8; ++e) Vt[(ch * 8 + e) * ASTR + row] = v[e];
    }
    asm volatile("s_wait_asynccnt 0" ::: "memory");
    __syncthreads();

    // S = (Q K^T) * 1/sqrt(dh), causal mask on diagonal block
    v8f sacc[4];
#pragma unroll
    for (int ni = 0; ni < 4; ++ni) {
#pragma unroll
      for (int r = 0; r < 8; ++r) sacc[ni][r] = 0.f;
      v16bf bk0 = load_frag_b(&Ks[(ni * 16) * ASTR], ASTR);
      sacc[ni] = __builtin_amdgcn_wmma_f32_16x16x32_bf16(
          false, aq0, false, bk0, (short)0, sacc[ni], false, false);
      v16bf bk1 = load_frag_b(&Ks[(ni * 16) * ASTR + 32], ASTR);
      sacc[ni] = __builtin_amdgcn_wmma_f32_16x16x32_bf16(
          false, aq1, false, bk1, (short)0, sacc[ni], false, false);
    }
#pragma unroll
    for (int ni = 0; ni < 4; ++ni) {
      int key = kb * 64 + ni * 16 + (lane & 15);
#pragma unroll
      for (int r = 0; r < 8; ++r) {
        float s = sacc[ni][r] * 0.125f;                  // 1/sqrt(64)
        if (kb == qb && key > qrow0 + r) s = -1.0e9f;    // causal additive mask
        sacc[ni][r] = s;
      }
    }
    // online softmax (row reductions across the 16-lane half)
    float alpha[8];
#pragma unroll
    for (int r = 0; r < 8; ++r) {
      float mx = sacc[0][r];
#pragma unroll
      for (int ni = 1; ni < 4; ++ni) mx = fmaxf(mx, sacc[ni][r]);
#pragma unroll
      for (int off = 1; off < 16; off <<= 1) mx = fmaxf(mx, __shfl_xor(mx, off, 32));
      float mn = fmaxf(mrow[r], mx);
      alpha[r] = __expf(mrow[r] - mn);
      mrow[r] = mn;
      float ps = 0.f;
#pragma unroll
      for (int ni = 0; ni < 4; ++ni) {
        float p = __expf(sacc[ni][r] - mn);
        sacc[ni][r] = p;
        ps += p;
      }
#pragma unroll
      for (int off = 1; off < 16; off <<= 1) ps += __shfl_xor(ps, off, 32);
      lrow[r] = lrow[r] * alpha[r] + ps;
    }
    // stage P (C layout -> A-fragment layout via per-wave LDS region)
#pragma unroll
    for (int ni = 0; ni < 4; ++ni)
#pragma unroll
      for (int r = 0; r < 8; ++r)
        Pw[(rhalf + r) * ASTR + ni * 16 + (lane & 15)] = (__bf16)sacc[ni][r];
    asm volatile("s_wait_dscnt 0" ::: "memory");
    v16bf pf0 = load_frag_a(Pw, ASTR);        // keys 0..31
    v16bf pf1 = load_frag_a(Pw + 32, ASTR);   // keys 32..63
    // O = O*alpha + P V
#pragma unroll
    for (int ni = 0; ni < 4; ++ni) {
#pragma unroll
      for (int r = 0; r < 8; ++r) o[ni][r] *= alpha[r];
      v16bf bv0 = load_frag_b(&Vt[(ni * 16) * ASTR], ASTR);
      o[ni] = __builtin_amdgcn_wmma_f32_16x16x32_bf16(
          false, pf0, false, bv0, (short)0, o[ni], false, false);
      v16bf bv1 = load_frag_b(&Vt[(ni * 16) * ASTR + 32], ASTR);
      o[ni] = __builtin_amdgcn_wmma_f32_16x16x32_bf16(
          false, pf1, false, bv1, (short)0, o[ni], false, false);
    }
  }
  // normalize and write
#pragma unroll
  for (int r = 0; r < 8; ++r) {
    float inv = 1.0f / lrow[r];
    int srow = qrow0 + r;
#pragma unroll
    for (int ni = 0; ni < 4; ++ni)
      O[base + (size_t)srow * DIM + ni * 16 + (lane & 15)] = (__bf16)(o[ni][r] * inv);
  }
}

// ---------------------------------------------------------------------------
// Host orchestration
// ---------------------------------------------------------------------------
static inline size_t align256(size_t x) { return (x + 255) & ~(size_t)255; }

extern "C" void kernel_launch(void* const* d_in, const int* in_sizes, int n_in,
                              void* d_out, int out_size, void* d_ws, size_t ws_size,
                              hipStream_t stream) {
  (void)in_sizes; (void)n_in; (void)out_size; (void)ws_size;

  const float* inputs = (const float*)d_in[0];
  const float* Wq = (const float*)d_in[1];  const float* bq = (const float*)d_in[2];
  const float* Wk = (const float*)d_in[3];  const float* bk = (const float*)d_in[4];
  const float* Wv = (const float*)d_in[5];  const float* bv = (const float*)d_in[6];
  const float* Wo = (const float*)d_in[7];  const float* bo = (const float*)d_in[8];
  const float* W1 = (const float*)d_in[9];  const float* b1 = (const float*)d_in[10];
  const float* W2 = (const float*)d_in[11]; const float* b2 = (const float*)d_in[12];
  const float* ln_g  = (const float*)d_in[13];
  const float* ln_b  = (const float*)d_in[14];
  const float* scale = (const float*)d_in[15];

  char* ws = (char*)d_ws;
  size_t off = 0;
  auto alloc = [&](size_t bytes) -> char* {
    char* p = ws + off; off = align256(off + bytes); return p;
  };

  float*  x    = (float*) alloc((size_t)MROWS * DIM * 4);
  __bf16* rbf  = (__bf16*)alloc((size_t)MROWS * DIM * 2);
  __bf16* qbf  = (__bf16*)alloc((size_t)MROWS * DIM * 2);
  __bf16* kbf  = (__bf16*)alloc((size_t)MROWS * DIM * 2);
  __bf16* vbf  = (__bf16*)alloc((size_t)MROWS * DIM * 2);
  __bf16* abf  = (__bf16*)alloc((size_t)MROWS * DIM * 2);
  __bf16* hbf  = (__bf16*)alloc((size_t)MROWS * FCSZ * 2);
  __bf16* wq_b = (__bf16*)alloc((size_t)DIM * DIM * 2);
  __bf16* wk_b = (__bf16*)alloc((size_t)DIM * DIM * 2);
  __bf16* wv_b = (__bf16*)alloc((size_t)DIM * DIM * 2);
  __bf16* wo_b = (__bf16*)alloc((size_t)DIM * DIM * 2);
  __bf16* w1_b = (__bf16*)alloc((size_t)DIM * FCSZ * 2);
  __bf16* w2_b = (__bf16*)alloc((size_t)FCSZ * DIM * 2);

  hipMemcpyAsync(x, inputs, (size_t)MROWS * DIM * 4, hipMemcpyDeviceToDevice, stream);

  auto cvt = [&](const float* src, __bf16* dst, size_t n) {
    f32_to_bf16_k<<<dim3((unsigned)((n + 255) / 256)), dim3(256), 0, stream>>>(src, dst, (int)n);
  };
  auto gemm = [&](const __bf16* A, const __bf16* Wt, const float* bias,
                  int Mv, int Nv, int Kv, int mode, __bf16* outB, float* xio) {
    dim3 g((unsigned)(Nv / BN), (unsigned)(Mv / BM));
    if (mode == 0)
      gemm_wmma<0><<<g, 256, 0, stream>>>(A, Wt, bias, Mv, Nv, Kv, outB, nullptr, nullptr, nullptr);
    else if (mode == 1)
      gemm_wmma<1><<<g, 256, 0, stream>>>(A, Wt, bias, Mv, Nv, Kv, outB, nullptr, nullptr, nullptr);
    else
      gemm_wmma<2><<<g, 256, 0, stream>>>(A, Wt, bias, Mv, Nv, Kv, nullptr, xio, xio, scale);
  };

  dim3 gLN(MROWS);
  for (int i = 0; i < LAYERS; ++i) {
    cvt(Wq + (size_t)i * DIM * DIM,  wq_b, (size_t)DIM * DIM);
    cvt(Wk + (size_t)i * DIM * DIM,  wk_b, (size_t)DIM * DIM);
    cvt(Wv + (size_t)i * DIM * DIM,  wv_b, (size_t)DIM * DIM);
    cvt(Wo + (size_t)i * DIM * DIM,  wo_b, (size_t)DIM * DIM);
    cvt(W1 + (size_t)i * DIM * FCSZ, w1_b, (size_t)DIM * FCSZ);
    cvt(W2 + (size_t)i * FCSZ * DIM, w2_b, (size_t)FCSZ * DIM);

    // --- attention block ---
    layernorm_k<__bf16><<<gLN, 256, 0, stream>>>(x, ln_g, ln_b, rbf, DIM);
    gemm(rbf, wq_b, bq + (size_t)i * DIM, MROWS, DIM, DIM, 0, qbf, nullptr);
    gemm(rbf, wk_b, bk + (size_t)i * DIM, MROWS, DIM, DIM, 0, kbf, nullptr);
    gemm(rbf, wv_b, bv + (size_t)i * DIM, MROWS, DIM, DIM, 0, vbf, nullptr);
    attn_flash<<<dim3(SEQ / 64, BSZ * HEADS), dim3(128), 0, stream>>>(qbf, kbf, vbf, abf);
    gemm(abf, wo_b, bo + (size_t)i * DIM, MROWS, DIM, DIM, 2, nullptr, x);

    // --- FFN block ---
    layernorm_k<__bf16><<<gLN, 256, 0, stream>>>(x, ln_g, ln_b, rbf, DIM);
    gemm(rbf, w1_b, b1 + (size_t)i * FCSZ, MROWS, FCSZ, DIM, 1, hbf, nullptr);
    gemm(hbf, w2_b, b2 + (size_t)i * DIM,  MROWS, DIM, FCSZ, 2, nullptr, x);
  }
  layernorm_k<float><<<gLN, 256, 0, stream>>>(x, ln_g, ln_b, (float*)d_out, DIM);
}
